// GNS_72275709657170
// MI455X (gfx1250) — matrix operand order repaired and checked by reading, not compiled, not run on hardware
//
#include <hip/hip_runtime.h>
#include <hip/hip_bf16.h>

typedef __attribute__((ext_vector_type(16))) _Float16 v16h;
typedef __attribute__((ext_vector_type(8)))  float    v8f;
typedef __attribute__((ext_vector_type(4)))  float    v4f;

// f16 weight region layout (offsets in halves), all transposed to Wt[n][Kpad]
constexpr int OFF_ENC_W1 = 0;       // 128 x 32   (K=8  pad 32)
constexpr int OFF_ENC_W2 = 4096;    // 32  x 128
constexpr int OFF_MSG_W1 = 8192;    // 2 x 128 x 96  (K=67 pad 96)
constexpr int OFF_MSG_W2 = 32768;   // 2 x 32  x 128
constexpr int OFF_UPD_W1 = 40960;   // 2 x 128 x 64
constexpr int OFF_UPD_W2 = 57344;   // 2 x 32  x 128
constexpr int OFF_DEC_W1 = 65536;   // 128 x 32
constexpr int OFF_DEC_W2 = 69632;   // 16  x 128  (N=4 pad 16)
constexpr int WT_TOTAL   = 71680;

__device__ __forceinline__ float eluf(float x) {
    return x > 0.f ? x : __expf(x) - 1.f;
}

union AFrag { v16h v; v4f q[2]; };

// A fragment (16x32 f16, MxK) from LDS row-major [16][Kpad].
// Lane L: row m=L&15, k-group kg=L>>4. Halves i<8 -> K=kb*32+kg*8+i,
// i>=8 -> K=kb*32+16+kg*8+(i-8)  (per ISA 16-bit A layout).
__device__ __forceinline__ v16h load_a_lds(const _Float16* base, int Kpad, int kb, int lane) {
    int m = lane & 15, kg = (lane >> 4) & 1;
    const _Float16* p = base + m * Kpad + kb * 32 + kg * 8;
    AFrag a;
    a.q[0] = *(const v4f*)(p);
    a.q[1] = *(const v4f*)(p + 16);
    return a.v;
}

// B fragment (32x16 f16, KxN) from global transposed weights Wt[n][Kpad]:
// lane L holds col n=L&15, i<8 -> K=kb*32+kg*8+i, i>=8 -> K=kb*32+16+kg*8+(i-8) (b128 loads).
__device__ __forceinline__ v16h load_b_glb(const _Float16* __restrict__ Wt, int Kpad,
                                           int kb, int tile, int lane) {
    int n = tile * 16 + (lane & 15), kg = (lane >> 4) & 1;
    const _Float16* p = Wt + n * Kpad + kb * 32 + kg * 8;
    AFrag b;
    b.q[0] = *(const v4f*)(p);
    b.q[1] = *(const v4f*)(p + 16);
    return b.v;
}

// GEMM1: sIn[16][KPAD] @ W1t -> hidden 128, fused bias+ELU, store f16 to sHid[16][128]
template <int KPAD>
__device__ __forceinline__ void mlp_hidden(const _Float16* sIn, const _Float16* __restrict__ W1t,
                                           const float* __restrict__ b1, _Float16* sHid, int lane) {
    constexpr int KB = KPAD / 32;
    v16h a[KB];
#pragma unroll
    for (int kb = 0; kb < KB; ++kb) a[kb] = load_a_lds(sIn, KPAD, kb, lane);
    int nl = lane & 15, kg = (lane >> 4) & 1;
#pragma unroll
    for (int t = 0; t < 8; ++t) {
        v8f c = {0.f, 0.f, 0.f, 0.f, 0.f, 0.f, 0.f, 0.f};
#pragma unroll
        for (int kb = 0; kb < KB; ++kb) {
            v16h b = load_b_glb(W1t, KPAD, kb, t, lane);
            c = __builtin_amdgcn_wmma_f32_16x16x32_f16(false, a[kb], false, b, (short)0, c,
                                                       false, false);
        }
        int n = t * 16 + nl;
        float bias = b1[n];
#pragma unroll
        for (int v = 0; v < 8; ++v) {
            sHid[(v + 8 * kg) * 128 + n] = (_Float16)eluf(c[v] + bias);
        }
    }
}

// GEMM2 output tile: hidden[16][128] @ W2t (K=128)
__device__ __forceinline__ v8f out_tile(const v16h a[4], const _Float16* __restrict__ W2t,
                                        int t, int lane) {
    v8f c = {0.f, 0.f, 0.f, 0.f, 0.f, 0.f, 0.f, 0.f};
#pragma unroll
    for (int kb = 0; kb < 4; ++kb) {
        v16h b = load_b_glb(W2t, 128, kb, t, lane);
        c = __builtin_amdgcn_wmma_f32_16x16x32_f16(false, a[kb], false, b, (short)0, c,
                                                   false, false);
    }
    return c;
}

// ---------------- weight prep: f32 [K][N] -> f16 transposed+padded [Npad][Kpad]
__device__ void tpad(const float* __restrict__ W, _Float16* __restrict__ Wt,
                     int K, int N, int Kpad, int Npad, int tid, int nth) {
    int tot = Npad * Kpad;
    for (int i = tid; i < tot; i += nth) {
        int n = i / Kpad, k = i - n * Kpad;
        float v = (k < K && n < N) ? W[k * N + n] : 0.f;
        Wt[i] = (_Float16)v;
    }
}

__global__ __launch_bounds__(256) void k_prep(const float* encW1, const float* encW2,
                                              const float* msgW1, const float* msgW2,
                                              const float* updW1, const float* updW2,
                                              const float* decW1, const float* decW2,
                                              _Float16* wt) {
    int tid = blockIdx.x * blockDim.x + threadIdx.x;
    int nth = gridDim.x * blockDim.x;
    tpad(encW1, wt + OFF_ENC_W1, 8, 128, 32, 128, tid, nth);
    tpad(encW2, wt + OFF_ENC_W2, 128, 32, 128, 32, tid, nth);
#pragma unroll
    for (int l = 0; l < 2; ++l) {
        tpad(msgW1 + l * 67 * 128, wt + OFF_MSG_W1 + l * 12288, 67, 128, 96, 128, tid, nth);
        tpad(msgW2 + l * 128 * 32, wt + OFF_MSG_W2 + l * 4096, 128, 32, 128, 32, tid, nth);
        tpad(updW1 + l * 64 * 128, wt + OFF_UPD_W1 + l * 8192, 64, 128, 64, 128, tid, nth);
        tpad(updW2 + l * 128 * 32, wt + OFF_UPD_W2 + l * 4096, 128, 32, 128, 32, tid, nth);
    }
    tpad(decW1, wt + OFF_DEC_W1, 32, 128, 32, 128, tid, nth);
    tpad(decW2, wt + OFF_DEC_W2, 128, 4, 128, 16, tid, nth);
}

__global__ __launch_bounds__(256) void k_zero(float* __restrict__ p, int n) {
    int i = blockIdx.x * blockDim.x + threadIdx.x;
    int st = gridDim.x * blockDim.x;
    for (; i < n; i += st) p[i] = 0.f;
}

// ---------------- encoder: h = elu(mlp(x))
__global__ __launch_bounds__(256) void k_encoder(const float* __restrict__ x,
                                                 const _Float16* __restrict__ W1t,
                                                 const float* __restrict__ b1,
                                                 const _Float16* __restrict__ W2t,
                                                 const float* __restrict__ b2,
                                                 float* __restrict__ hout, int nN) {
    __shared__ __align__(16) _Float16 sIn[8][16 * 32];
    __shared__ __align__(16) _Float16 sHid[8][16 * 128];
    int wave = threadIdx.x >> 5, lane = threadIdx.x & 31;
    int tile = blockIdx.x * 8 + wave;
    int row = lane >> 1, half = lane & 1;
    int node = tile * 16 + row;
    bool valid = node < nN;
    _Float16* in = sIn[wave];
    if (half == 0) {
#pragma unroll
        for (int i = 0; i < 16; ++i) {
            float v = (valid && i < 8) ? x[node * 8 + i] : 0.f;
            in[row * 32 + i] = (_Float16)v;
        }
    } else {
#pragma unroll
        for (int i = 0; i < 16; ++i) in[row * 32 + 16 + i] = (_Float16)0.f;
    }
    __syncthreads();
    mlp_hidden<32>(in, W1t, b1, sHid[wave], lane);
    __syncthreads();
    v16h a[4];
#pragma unroll
    for (int kb = 0; kb < 4; ++kb) a[kb] = load_a_lds(sHid[wave], 128, kb, lane);
    int nl = lane & 15, kg = (lane >> 4) & 1;
#pragma unroll
    for (int t = 0; t < 2; ++t) {
        v8f c = out_tile(a, W2t, t, lane);
        int n = t * 16 + nl;
        float bias = b2[n];
#pragma unroll
        for (int v = 0; v < 8; ++v) {
            int nd = tile * 16 + v + 8 * kg;
            if (nd < nN) hout[nd * 32 + n] = eluf(c[v] + bias);
        }
    }
}

// ---------------- edge messages: m = mlp([h[dst],h[src],ea]); atomic scatter to aggr[dst]
__global__ __launch_bounds__(256) void k_edge(const float* __restrict__ h,
                                              const int* __restrict__ ei,
                                              const float* __restrict__ ea,
                                              const _Float16* __restrict__ W1t,
                                              const float* __restrict__ b1,
                                              const _Float16* __restrict__ W2t,
                                              const float* __restrict__ b2,
                                              float* __restrict__ aggr, int nE) {
    __shared__ __align__(16) _Float16 sIn[8][16 * 96];
    __shared__ __align__(16) _Float16 sHid[8][16 * 128];
    __shared__ int sDst[8][16];
    int wave = threadIdx.x >> 5, lane = threadIdx.x & 31;
    int tile = blockIdx.x * 8 + wave;
    int row = lane >> 1, half = lane & 1;
    int e = tile * 16 + row;
    bool valid = e < nE;
    int srcn = valid ? ei[e] : 0;
    int dstn = valid ? ei[nE + e] : 0;
    if (half == 0) sDst[wave][row] = valid ? dstn : -1;
    _Float16* in = sIn[wave];
    int rb = row * 96;
    if (half == 0) {
#pragma unroll
        for (int i = 0; i < 32; ++i) in[rb + i] = (_Float16)(valid ? h[dstn * 32 + i] : 0.f);
#pragma unroll
        for (int i = 0; i < 16; ++i) in[rb + 32 + i] = (_Float16)(valid ? h[srcn * 32 + i] : 0.f);
    } else {
#pragma unroll
        for (int i = 0; i < 16; ++i)
            in[rb + 48 + i] = (_Float16)(valid ? h[srcn * 32 + 16 + i] : 0.f);
#pragma unroll
        for (int i = 0; i < 3; ++i) in[rb + 64 + i] = (_Float16)(valid ? ea[e * 3 + i] : 0.f);
#pragma unroll
        for (int i = 0; i < 29; ++i) in[rb + 67 + i] = (_Float16)0.f;
    }
    __syncthreads();
    mlp_hidden<96>(in, W1t, b1, sHid[wave], lane);
    __syncthreads();
    v16h a[4];
#pragma unroll
    for (int kb = 0; kb < 4; ++kb) a[kb] = load_a_lds(sHid[wave], 128, kb, lane);
    int nl = lane & 15, kg = (lane >> 4) & 1;
#pragma unroll
    for (int t = 0; t < 2; ++t) {
        v8f c = out_tile(a, W2t, t, lane);
        int n = t * 16 + nl;
        float bias = b2[n];
#pragma unroll
        for (int v = 0; v < 8; ++v) {
            int d = sDst[wave][v + 8 * kg];
            if (d >= 0) atomicAdd(&aggr[d * 32 + n], c[v] + bias);
        }
    }
}

// ---------------- node update: h = groupnorm(elu(mlp([h,aggr])))
__global__ __launch_bounds__(256) void k_update(float* __restrict__ h,
                                                const float* __restrict__ aggr,
                                                const _Float16* __restrict__ W1t,
                                                const float* __restrict__ b1,
                                                const _Float16* __restrict__ W2t,
                                                const float* __restrict__ b2,
                                                const float* __restrict__ gamma,
                                                const float* __restrict__ beta, int nN) {
    __shared__ __align__(16) _Float16 sIn[8][16 * 64];
    __shared__ __align__(16) _Float16 sHid[8][16 * 128];
    int wave = threadIdx.x >> 5, lane = threadIdx.x & 31;
    int tile = blockIdx.x * 8 + wave;
    int row = lane >> 1, half = lane & 1;
    int node = tile * 16 + row;
    bool valid = node < nN;
    _Float16* in = sIn[wave];
    int rb = row * 64;
    if (half == 0) {
#pragma unroll
        for (int i = 0; i < 32; ++i) in[rb + i] = (_Float16)(valid ? h[node * 32 + i] : 0.f);
    } else {
#pragma unroll
        for (int i = 0; i < 32; ++i)
            in[rb + 32 + i] = (_Float16)(valid ? aggr[node * 32 + i] : 0.f);
    }
    __syncthreads();
    mlp_hidden<64>(in, W1t, b1, sHid[wave], lane);
    __syncthreads();
    v16h a[4];
#pragma unroll
    for (int kb = 0; kb < 4; ++kb) a[kb] = load_a_lds(sHid[wave], 128, kb, lane);
    int nl = lane & 15, kg = (lane >> 4) & 1;
    float* hn = (float*)sIn[wave];  // reuse 2KB region as [16][32] f32
#pragma unroll
    for (int t = 0; t < 2; ++t) {
        v8f c = out_tile(a, W2t, t, lane);
        int n = t * 16 + nl;
        float bias = b2[n];
#pragma unroll
        for (int v = 0; v < 8; ++v) hn[(v + 8 * kg) * 32 + n] = eluf(c[v] + bias);
    }
    __syncthreads();
    // GroupNorm(2 groups of 16 ch): lane -> (row = lane&15, group = lane>>4)
    int r = lane & 15, g = (lane >> 4) & 1;
    int nd = tile * 16 + r;
    const float* p = hn + r * 32 + g * 16;
    float mu = 0.f;
#pragma unroll
    for (int i = 0; i < 16; ++i) mu += p[i];
    mu *= (1.f / 16.f);
    float var = 0.f;
#pragma unroll
    for (int i = 0; i < 16; ++i) {
        float d = p[i] - mu;
        var += d * d;
    }
    var *= (1.f / 16.f);
    float rs = rsqrtf(var + 1e-5f);
    if (nd < nN) {
#pragma unroll
        for (int i = 0; i < 16; ++i) {
            int ch = g * 16 + i;
            h[nd * 32 + ch] = (p[i] - mu) * rs * gamma[ch] + beta[ch];
        }
    }
}

// ---------------- decoder: out = mlp(h) -> [N,4]
__global__ __launch_bounds__(256) void k_decoder(const float* __restrict__ h,
                                                 const _Float16* __restrict__ W1t,
                                                 const float* __restrict__ b1,
                                                 const _Float16* __restrict__ W2t,
                                                 const float* __restrict__ b2,
                                                 float* __restrict__ out, int nN) {
    __shared__ __align__(16) _Float16 sIn[8][16 * 32];
    __shared__ __align__(16) _Float16 sHid[8][16 * 128];
    int wave = threadIdx.x >> 5, lane = threadIdx.x & 31;
    int tile = blockIdx.x * 8 + wave;
    int row = lane >> 1, half = lane & 1;
    int node = tile * 16 + row;
    bool valid = node < nN;
    _Float16* in = sIn[wave];
#pragma unroll
    for (int i = 0; i < 16; ++i) {
        int f = half * 16 + i;
        in[row * 32 + f] = (_Float16)(valid ? h[node * 32 + f] : 0.f);
    }
    __syncthreads();
    mlp_hidden<32>(in, W1t, b1, sHid[wave], lane);
    __syncthreads();
    v16h a[4];
#pragma unroll
    for (int kb = 0; kb < 4; ++kb) a[kb] = load_a_lds(sHid[wave], 128, kb, lane);
    int nl = lane & 15, kg = (lane >> 4) & 1;
    v8f c = out_tile(a, W2t, 0, lane);
    if (nl < 4) {
        float bias = b2[nl];
#pragma unroll
        for (int v = 0; v < 8; ++v) {
            int nd = tile * 16 + v + 8 * kg;
            if (nd < nN) out[nd * 4 + nl] = c[v] + bias;
        }
    }
}

extern "C" void kernel_launch(void* const* d_in, const int* in_sizes, int n_in,
                              void* d_out, int out_size, void* d_ws, size_t ws_size,
                              hipStream_t stream) {
    const float* x     = (const float*)d_in[0];
    const int*   ei    = (const int*)d_in[1];
    const float* ea    = (const float*)d_in[2];
    const float* encW1 = (const float*)d_in[3];
    const float* encb1 = (const float*)d_in[4];
    const float* encW2 = (const float*)d_in[5];
    const float* encb2 = (const float*)d_in[6];
    const float* msgW1 = (const float*)d_in[7];
    const float* msgb1 = (const float*)d_in[8];
    const float* msgW2 = (const float*)d_in[9];
    const float* msgb2 = (const float*)d_in[10];
    const float* updW1 = (const float*)d_in[11];
    const float* updb1 = (const float*)d_in[12];
    const float* updW2 = (const float*)d_in[13];
    const float* updb2 = (const float*)d_in[14];
    const float* gamma = (const float*)d_in[15];
    const float* beta  = (const float*)d_in[16];
    const float* decW1 = (const float*)d_in[17];
    const float* decb1 = (const float*)d_in[18];
    const float* decW2 = (const float*)d_in[19];
    const float* decb2 = (const float*)d_in[20];

    int N = in_sizes[0] / 8;
    int E = in_sizes[2] / 3;

    float* h    = (float*)d_ws;
    float* aggr = h + (size_t)N * 32;
    _Float16* wt = (_Float16*)(aggr + (size_t)N * 32);

    k_prep<<<64, 256, 0, stream>>>(encW1, encW2, msgW1, msgW2, updW1, updW2, decW1, decW2, wt);

    int ntiles = (N + 15) / 16, nblk = (ntiles + 7) / 8;
    int etiles = (E + 15) / 16, eblk = (etiles + 7) / 8;

    k_encoder<<<nblk, 256, 0, stream>>>(x, wt + OFF_ENC_W1, encb1, wt + OFF_ENC_W2, encb2, h, N);

    for (int l = 0; l < 2; ++l) {
        k_zero<<<512, 256, 0, stream>>>(aggr, N * 32);
        k_edge<<<eblk, 256, 0, stream>>>(h, ei, ea,
                                         wt + OFF_MSG_W1 + l * 12288, msgb1 + l * 128,
                                         wt + OFF_MSG_W2 + l * 4096, msgb2 + l * 32, aggr, E);
        k_update<<<nblk, 256, 0, stream>>>(h, aggr,
                                           wt + OFF_UPD_W1 + l * 8192, updb1 + l * 128,
                                           wt + OFF_UPD_W2 + l * 4096, updb2 + l * 32,
                                           gamma, beta, N);
    }
    k_decoder<<<nblk, 256, 0, stream>>>(h, wt + OFF_DEC_W1, decb1, wt + OFF_DEC_W2, decb2,
                                        (float*)d_out, N);
}
